// Grok1DecoderLayer_850403524960
// MI455X (gfx1250) — matrix operand order repaired and checked by reading, not compiled, not run on hardware
//
#include <hip/hip_runtime.h>
#include <hip/hip_bf16.h>
#include <math.h>

// Problem dims (Grok1 decoder layer)
#define T_DIM  1024
#define D_DIM  2048
#define H_N    16
#define KH_N   4
#define HD_N   128
#define E_N    8
#define I_N    2048
#define QKV_N  ((H_N + 2*KH_N)*HD_N)   // 3072
#define GU_N   (2*I_N)                 // 4096
#define CAPV   30.0f

typedef __attribute__((ext_vector_type(16))) __bf16 v16bf;
typedef __attribute__((ext_vector_type(8)))  float  v8f;

union Frag { v16bf v; unsigned int u[8]; };

__device__ __forceinline__ unsigned short f2bf(float f) {
  unsigned int x = __float_as_uint(f);
  x += 0x7FFFu + ((x >> 16) & 1u);
  return (unsigned short)(x >> 16);
}

#define WMMA_BF16(A_, B_, C_) \
  __builtin_amdgcn_wmma_f32_16x16x32_bf16(false, (A_), false, (B_), (short)0, (C_), false, false)

// ---------------------------------------------------------------------------
// Generic bf16 WMMA GEMM: C[M,N] = A[M,K](bf16) x B[K,N](fp32->bf16 in LDS)
// MODE 0: store; MODE 1: C += scale[row*scaleStride + z] * acc (atomic)
// ---------------------------------------------------------------------------
#define BM 128
#define BN 128
#define BK 32
#define AST 34           // padded ushort row stride (17 dwords, odd -> no bank conflict)
#define AST32 (AST/2)    // 17

template <int MODE>
__global__ void __launch_bounds__(256)
k_gemm_bf16(const unsigned short* __restrict__ A, const float* __restrict__ B,
            float* __restrict__ C, int M, int N, int K,
            long long sAz, long long sBz, long long sCz,
            const float* __restrict__ scale, int scaleStride) {
  __shared__ unsigned short As[BM * AST];
  __shared__ unsigned short Bs[BN * AST];
  unsigned int* As32 = (unsigned int*)As;
  unsigned int* Bs32 = (unsigned int*)Bs;
  const int z = blockIdx.z;
  A += (long long)z * sAz; B += (long long)z * sBz; C += (long long)z * sCz;
  const int m0 = blockIdx.y * BM, n0 = blockIdx.x * BN;
  const int tid = threadIdx.x, w = tid >> 5, l = tid & 31;
  const int wm = (w & 3) * 32;    // wave sub-tile row
  const int wn = (w >> 2) * 64;   // wave sub-tile col

  v8f acc[2][4];
#pragma unroll
  for (int i = 0; i < 2; i++)
#pragma unroll
    for (int j = 0; j < 4; j++) acc[i][j] = (v8f)0.0f;

  // staging coordinates (fixed per thread)
  const int arow = tid >> 1, aseg = (tid & 1) * 16;          // A: 16 bf16 / thread
  const int bkr  = (tid >> 4) * 2, bnc = (tid & 15) * 8;     // B: 2 k-rows x 8 n / thread

  for (int k0 = 0; k0 < K; k0 += BK) {
    { // stage A tile (bf16 128x32) -> packed b32 LDS stores
      const unsigned short* src = A + (long long)(m0 + arow) * K + k0 + aseg;
      uint4 d0 = *(const uint4*)(src);
      uint4 d1 = *(const uint4*)(src + 8);
      unsigned int tmp[8] = {d0.x, d0.y, d0.z, d0.w, d1.x, d1.y, d1.z, d1.w};
#pragma unroll
      for (int i = 0; i < 8; i++)
        As32[arow * AST32 + (aseg >> 1) + i] = tmp[i];
    }
    { // stage B tile (fp32 32x128 -> bf16 LDS [n][k]), k-pairs packed per u32
      const float* s0 = B + (long long)(k0 + bkr) * N + n0 + bnc;
      const float* s1 = s0 + N;
      float4 a0 = *(const float4*)s0, a1 = *(const float4*)(s0 + 4);
      float4 b0 = *(const float4*)s1, b1 = *(const float4*)(s1 + 4);
      float r0[8] = {a0.x, a0.y, a0.z, a0.w, a1.x, a1.y, a1.z, a1.w};
      float r1[8] = {b0.x, b0.y, b0.z, b0.w, b1.x, b1.y, b1.z, b1.w};
#pragma unroll
      for (int i = 0; i < 8; i++)
        Bs32[(bnc + i) * AST32 + (bkr >> 1)] =
            (unsigned int)f2bf(r0[i]) | ((unsigned int)f2bf(r1[i]) << 16);
      // prefetch next B k-slab (HBM-bound on fp32 weights)
      if (k0 + BK < K) __builtin_prefetch(s0 + (long long)BK * N, 0, 0);
    }
    __syncthreads();

    Frag fa[2], fb[4];
#pragma unroll
    for (int i = 0; i < 2; i++) {
      int m = wm + i * 16 + (l & 15);
#pragma unroll
      for (int v = 0; v < 8; v++) {
        int kb = ((v >= 4) ? 16 : 0) + ((l >= 16) ? 8 : 0) + (v & 3) * 2;
        fa[i].u[v] = *(const unsigned int*)&As[m * AST + kb];
      }
    }
#pragma unroll
    for (int j = 0; j < 4; j++) {
      int n = wn + j * 16 + (l & 15);
      int kh = (l >= 16) ? 8 : 0;
#pragma unroll
      for (int v = 0; v < 8; v++)
        fb[j].u[v] = *(const unsigned int*)&Bs[n * AST + ((v >= 4) ? 16 : 0) + kh + (v & 3) * 2];
    }
#pragma unroll
    for (int i = 0; i < 2; i++)
#pragma unroll
      for (int j = 0; j < 4; j++)
        acc[i][j] = WMMA_BF16(fa[i].v, fb[j].v, acc[i][j]);
    __syncthreads();
  }

  // epilogue: hoisted row base + scale, immediate-offset stores
#pragma unroll
  for (int i = 0; i < 2; i++)
#pragma unroll
    for (int r = 0; r < 8; r++) {
      int row = m0 + wm + i * 16 + r + 8 * (l >> 4);   // C layout: VGPR r
      float* crow = C + (long long)row * N + n0 + wn + (l & 15);
      float sv = (MODE == 1) ? scale[(long long)row * scaleStride + z] : 0.f;
#pragma unroll
      for (int j = 0; j < 4; j++) {
        float v = acc[i][j][r];
        if (MODE == 0) crow[j * 16] = v;
        else           atomicAdd(&crow[j * 16], sv * v);
      }
    }
}

// ---------------------------------------------------------------------------
// RMSNorm: y = x * rsqrt(mean(x^2)+eps) * w  (fp32 and/or bf16 outputs)
// ---------------------------------------------------------------------------
__global__ void __launch_bounds__(256)
k_rmsnorm(const float* __restrict__ x, const float* __restrict__ w,
          float* __restrict__ y32, unsigned short* __restrict__ y16) {
  __shared__ float red[8];
  long long row = blockIdx.x;
  const float* xr = x + row * D_DIM;
  float ss = 0.f;
  for (int d = threadIdx.x; d < D_DIM; d += 256) { float v = xr[d]; ss += v * v; }
#pragma unroll
  for (int m = 16; m >= 1; m >>= 1) ss += __shfl_xor(ss, m, 32);
  if ((threadIdx.x & 31) == 0) red[threadIdx.x >> 5] = ss;
  __syncthreads();
  float tot = 0.f;
#pragma unroll
  for (int i = 0; i < 8; i++) tot += red[i];
  float r = rsqrtf(tot / (float)D_DIM + 1e-5f);
  for (int d = threadIdx.x; d < D_DIM; d += 256) {
    float v = xr[d] * r * w[d];
    if (y32) y32[row * D_DIM + d] = v;
    if (y16) y16[row * D_DIM + d] = f2bf(v);
  }
}

// out = resid + rmsnorm(a, w)
__global__ void __launch_bounds__(256)
k_add_rmsnorm(const float* __restrict__ a, const float* __restrict__ w,
              const float* __restrict__ resid, float* __restrict__ out) {
  __shared__ float red[8];
  long long row = blockIdx.x;
  const float* xr = a + row * D_DIM;
  float ss = 0.f;
  for (int d = threadIdx.x; d < D_DIM; d += 256) { float v = xr[d]; ss += v * v; }
#pragma unroll
  for (int m = 16; m >= 1; m >>= 1) ss += __shfl_xor(ss, m, 32);
  if ((threadIdx.x & 31) == 0) red[threadIdx.x >> 5] = ss;
  __syncthreads();
  float tot = 0.f;
#pragma unroll
  for (int i = 0; i < 8; i++) tot += red[i];
  float r = rsqrtf(tot / (float)D_DIM + 1e-5f);
  for (int d = threadIdx.x; d < D_DIM; d += 256)
    out[row * D_DIM + d] = resid[row * D_DIM + d] + xr[d] * r * w[d];
}

// ---------------------------------------------------------------------------
// NeoX RoPE on q/k + bf16 repack of q/k/v from the qkv GEMM output
// ---------------------------------------------------------------------------
__global__ void __launch_bounds__(256)
k_rope(const int* __restrict__ positions, const float* __restrict__ qkv,
       unsigned short* __restrict__ qb, unsigned short* __restrict__ kb,
       unsigned short* __restrict__ vb) {
  const int t = blockIdx.x;
  const float pos = (float)positions[t];
  const float* row = qkv + (long long)t * QKV_N;
  for (int p = threadIdx.x; p < H_N * 64; p += 256) {
    int hh = p >> 6, i = p & 63;
    float inv = powf(10000.0f, -(float)i / 64.0f);
    float f = pos * inv, c = cosf(f), s = sinf(f);
    float x1 = row[hh * HD_N + i], x2 = row[hh * HD_N + 64 + i];
    long long ob = ((long long)t * H_N + hh) * HD_N;
    qb[ob + i]      = f2bf(x1 * c - x2 * s);
    qb[ob + 64 + i] = f2bf(x2 * c + x1 * s);
  }
  if (threadIdx.x < KH_N * 64) {
    int hh = threadIdx.x >> 6, i = threadIdx.x & 63;
    float inv = powf(10000.0f, -(float)i / 64.0f);
    float f = pos * inv, c = cosf(f), s = sinf(f);
    const float* kr = row + H_N * HD_N;
    float x1 = kr[hh * HD_N + i], x2 = kr[hh * HD_N + 64 + i];
    long long ob = ((long long)t * KH_N + hh) * HD_N;
    kb[ob + i]      = f2bf(x1 * c - x2 * s);
    kb[ob + 64 + i] = f2bf(x2 * c + x1 * s);
  }
  const float* vr = row + (H_N + KH_N) * HD_N;
  for (int j = threadIdx.x; j < KH_N * HD_N; j += 256)
    vb[(long long)t * KH_N * HD_N + j] = f2bf(vr[j]);
}

// ---------------------------------------------------------------------------
// Flash-style causal attention with tanh softcap. One block = (head, 128 q rows).
// Each wave owns 16 q rows -> softmax stats are wave-private.
// ---------------------------------------------------------------------------
#define QT 128
#define KT 64
#define QST (HD_N + 2)   // 130 ushorts = 65 dwords (odd -> conflict-free)
#define QST32 (QST/2)    // 65
#define VST (KT + 2)     // 66 ushorts = 33 dwords
#define VST32 (VST/2)    // 33

__global__ void __launch_bounds__(256)
k_attn(const unsigned short* __restrict__ qb, const unsigned short* __restrict__ kb,
       const unsigned short* __restrict__ vb, unsigned short* __restrict__ ob) {
  __shared__ unsigned short Qs[QT * QST];
  __shared__ unsigned short Ks[KT * QST];   // row-major == B-layout [n=key][k=hd]
  __shared__ unsigned short Vt[HD_N * VST]; // transposed V: [n=hd][k=key]
  __shared__ unsigned short Ps[QT * VST];   // P as A-matrix [m=q][k=key]
  unsigned int* Qs32 = (unsigned int*)Qs;
  unsigned int* Ks32 = (unsigned int*)Ks;
  unsigned int* Vt32 = (unsigned int*)Vt;

  const int h = blockIdx.y, kv = h >> 2;
  const int q0 = blockIdx.x * QT;
  const int tid = threadIdx.x, w = tid >> 5, l = tid & 31;

  { // stage Q (128x128 bf16): 64 elems/thread, packed b32 stores
    int row = tid >> 1, seg = (tid & 1) * 64;
    const unsigned short* src = qb + ((long long)(q0 + row) * H_N + h) * HD_N + seg;
#pragma unroll
    for (int i = 0; i < 64; i += 8) {
      uint4 d = *(const uint4*)(src + i);
      unsigned int t4[4] = {d.x, d.y, d.z, d.w};
#pragma unroll
      for (int p = 0; p < 4; p++)
        Qs32[row * QST32 + ((seg + i) >> 1) + p] = t4[p];
    }
  }

  float mrow[8], lrow[8];
  v8f o[8];
#pragma unroll
  for (int r = 0; r < 8; r++) { mrow[r] = -3.0e38f; lrow[r] = 0.f; }
#pragma unroll
  for (int j = 0; j < 8; j++) o[j] = (v8f)0.0f;

  const float sc = 0.08838834764831845f;  // HD^-0.5
  const int ktiles = (q0 + QT) / KT;

  for (int kt = 0; kt < ktiles; kt++) {
    const int kbase = kt * KT;
    __syncthreads();  // protects Ks/Vt reuse (also covers Q staging on iter 0)
    { // stage K tile 64x128, packed b32 stores
      int row = tid >> 2, seg = (tid & 3) * 32;
      const unsigned short* src = kb + ((long long)(kbase + row) * KH_N + kv) * HD_N + seg;
#pragma unroll
      for (int i = 0; i < 32; i += 8) {
        uint4 d = *(const uint4*)(src + i);
        unsigned int t4[4] = {d.x, d.y, d.z, d.w};
#pragma unroll
        for (int p = 0; p < 4; p++)
          Ks32[row * QST32 + ((seg + i) >> 1) + p] = t4[p];
      }
    }
    { // stage V transposed: Vt[hd][key]; key-pair per thread -> packed b32 stores
      int pr = tid >> 3;              // key pair index 0..31
      int seg = (tid & 7) * 16;       // hd segment
      const unsigned short* s0 = vb + ((long long)(kbase + 2 * pr) * KH_N + kv) * HD_N + seg;
      const unsigned short* s1 = s0 + (long long)KH_N * HD_N;
      uint4 a0 = *(const uint4*)s0, a1 = *(const uint4*)(s0 + 8);
      uint4 b0 = *(const uint4*)s1, b1 = *(const uint4*)(s1 + 8);
      unsigned int ua[8] = {a0.x, a0.y, a0.z, a0.w, a1.x, a1.y, a1.z, a1.w};
      unsigned int ub[8] = {b0.x, b0.y, b0.z, b0.w, b1.x, b1.y, b1.z, b1.w};
#pragma unroll
      for (int i = 0; i < 8; i++) {
        unsigned int x0 = ua[i], x1 = ub[i];
        Vt32[(seg + 2*i    ) * VST32 + pr] = (x0 & 0xFFFFu)  | (x1 << 16);
        Vt32[(seg + 2*i + 1) * VST32 + pr] = (x0 >> 16)      | (x1 & 0xFFFF0000u);
      }
    }
    __syncthreads();

    // S = Q K^T for this wave's 16 rows (K-dim = 128 -> 4 WMMA steps)
    v8f s[4];
#pragma unroll
    for (int j = 0; j < 4; j++) s[j] = (v8f)0.0f;
#pragma unroll
    for (int ks = 0; ks < 4; ks++) {
      Frag fa;
      int m = 16 * w + (l & 15);
#pragma unroll
      for (int v = 0; v < 8; v++) {
        int kb2 = ks * 32 + ((v >= 4) ? 16 : 0) + ((l >= 16) ? 8 : 0) + (v & 3) * 2;
        fa.u[v] = *(const unsigned int*)&Qs[m * QST + kb2];
      }
#pragma unroll
      for (int j = 0; j < 4; j++) {
        Frag fb;
        int n = j * 16 + (l & 15);
        int kh = ks * 32 + ((l >= 16) ? 8 : 0);
#pragma unroll
        for (int v = 0; v < 8; v++)
          fb.u[v] = *(const unsigned int*)&Ks[n * QST + kh + ((v >= 4) ? 16 : 0) + (v & 3) * 2];
        s[j] = WMMA_BF16(fa.v, fb.v, s[j]);
      }
    }

    // scale, tanh softcap, causal mask
#pragma unroll
    for (int j = 0; j < 4; j++)
#pragma unroll
      for (int r = 0; r < 8; r++) {
        float x = s[j][r] * sc;
        x = CAPV * tanhf(x / CAPV);
        int qg = q0 + 16 * w + r + 8 * (l >> 4);
        int kg = kbase + j * 16 + (l & 15);
        s[j][r] = (kg > qg) ? -1e30f : x;
      }

    // online softmax (rows live in 16-lane halves; xor-reduce stays inside)
#pragma unroll
    for (int r = 0; r < 8; r++) {
      float mx = s[0][r];
#pragma unroll
      for (int j = 1; j < 4; j++) mx = fmaxf(mx, s[j][r]);
#pragma unroll
      for (int d = 1; d < 16; d <<= 1) mx = fmaxf(mx, __shfl_xor(mx, d, 32));
      float mnew = fmaxf(mrow[r], mx);
      float rs = 0.f;
#pragma unroll
      for (int j = 0; j < 4; j++) { float e = __expf(s[j][r] - mnew); s[j][r] = e; rs += e; }
#pragma unroll
      for (int d = 1; d < 16; d <<= 1) rs += __shfl_xor(rs, d, 32);
      float alpha = __expf(mrow[r] - mnew);
      lrow[r] = lrow[r] * alpha + rs;
      mrow[r] = mnew;
#pragma unroll
      for (int j = 0; j < 8; j++) o[j][r] *= alpha;
    }

    // P -> LDS (A-layout, wave-private rows; same-wave LDS ops are in-order)
#pragma unroll
    for (int j = 0; j < 4; j++)
#pragma unroll
      for (int r = 0; r < 8; r++)
        Ps[(16 * w + r + 8 * (l >> 4)) * VST + j * 16 + (l & 15)] = f2bf(s[j][r]);

    // O += P * V  (K-dim = 64 -> 2 WMMA steps, N = 128 -> 8 frags)
#pragma unroll
    for (int ks = 0; ks < 2; ks++) {
      Frag fa;
      int m = 16 * w + (l & 15);
#pragma unroll
      for (int v = 0; v < 8; v++) {
        int kb2 = ks * 32 + ((v >= 4) ? 16 : 0) + ((l >= 16) ? 8 : 0) + (v & 3) * 2;
        fa.u[v] = *(const unsigned int*)&Ps[m * VST + kb2];
      }
#pragma unroll
      for (int j = 0; j < 8; j++) {
        Frag fb;
        int n = j * 16 + (l & 15);
        int kh = ks * 32 + ((l >= 16) ? 8 : 0);
#pragma unroll
        for (int v = 0; v < 8; v++)
          fb.u[v] = *(const unsigned int*)&Vt[n * VST + kh + ((v >= 4) ? 16 : 0) + (v & 3) * 2];
        o[j] = WMMA_BF16(fa.v, fb.v, o[j]);
      }
    }
  }

  // epilogue: O / l, write bf16 for the w_o GEMM
#pragma unroll
  for (int j = 0; j < 8; j++)
#pragma unroll
    for (int r = 0; r < 8; r++) {
      int qg = q0 + 16 * w + r + 8 * (l >> 4);
      int col = j * 16 + (l & 15);
      ob[(long long)qg * (H_N * HD_N) + h * HD_N + col] = f2bf(o[j][r] / lrow[r]);
    }
}

// ---------------------------------------------------------------------------
// MoE router: logits -> softcap -> softmax -> top-2 -> dense comb[T,E]
// ---------------------------------------------------------------------------
__global__ void __launch_bounds__(32)
k_gate(const float* __restrict__ h, const float* __restrict__ wg,
       float* __restrict__ comb) {
  const int t = blockIdx.x, l = threadIdx.x;
  float a[E_N];
#pragma unroll
  for (int e = 0; e < E_N; e++) a[e] = 0.f;
  const float* xr = h + (long long)t * D_DIM;
  for (int d = l; d < D_DIM; d += 32) {
    float xv = xr[d];
    const float* wr = wg + (long long)d * E_N;
#pragma unroll
    for (int e = 0; e < E_N; e++) a[e] += xv * wr[e];
  }
#pragma unroll
  for (int e = 0; e < E_N; e++)
#pragma unroll
    for (int m = 16; m >= 1; m >>= 1) a[e] += __shfl_xor(a[e], m, 32);
  if (l == 0) {
    float p[E_N], mx = -3.0e38f;
#pragma unroll
    for (int e = 0; e < E_N; e++) { a[e] = CAPV * tanhf(a[e] / CAPV); mx = fmaxf(mx, a[e]); }
    float sum = 0.f;
#pragma unroll
    for (int e = 0; e < E_N; e++) { p[e] = __expf(a[e] - mx); sum += p[e]; }
#pragma unroll
    for (int e = 0; e < E_N; e++) p[e] /= sum;
    int i1 = 0;
#pragma unroll
    for (int e = 1; e < E_N; e++) if (p[e] > p[i1]) i1 = e;
    int i2 = (i1 == 0) ? 1 : 0;
#pragma unroll
    for (int e = 0; e < E_N; e++) if (e != i1 && p[e] > p[i2]) i2 = e;
#pragma unroll
    for (int e = 0; e < E_N; e++)
      comb[(long long)t * E_N + e] = (e == i1 || e == i2) ? p[e] : 0.f;
  }
}

// act = gelu_tanh(g) * u (bf16 output for the down-GEMM A matrix)
__global__ void __launch_bounds__(256)
k_act(const float* __restrict__ gu, unsigned short* __restrict__ act) {
  long long idx = (long long)blockIdx.x * 256 + threadIdx.x;
  long long et = idx / I_N;   // e*T + t
  int i = (int)(idx % I_N);
  float g = gu[et * GU_N + i];
  float u = gu[et * GU_N + I_N + i];
  float g3 = g * g * g;
  float ge = 0.5f * g * (1.0f + tanhf(0.7978845608028654f * (g + 0.044715f * g3)));
  act[et * I_N + i] = f2bf(ge * u);
}

__global__ void __launch_bounds__(256)
k_zero(float* __restrict__ p) {
  p[(long long)blockIdx.x * 256 + threadIdx.x] = 0.f;
}

// ---------------------------------------------------------------------------
extern "C" void kernel_launch(void* const* d_in, const int* in_sizes, int n_in,
                              void* d_out, int out_size, void* d_ws, size_t ws_size,
                              hipStream_t stream) {
  const int*   positions = (const int*)  d_in[0];
  const float* hidden    = (const float*)d_in[1];
  const float* w_qkv     = (const float*)d_in[2];
  const float* w_o       = (const float*)d_in[3];
  const float* w_gate    = (const float*)d_in[4];
  const float* w_gate_up = (const float*)d_in[5];
  const float* w_down    = (const float*)d_in[6];
  const float* w_pre_attn  = (const float*)d_in[7];
  const float* w_post_attn = (const float*)d_in[8];
  const float* w_pre_moe   = (const float*)d_in[9];
  const float* w_post_moe  = (const float*)d_in[10];
  float* out = (float*)d_out;

  // workspace carve-up (256B aligned)
  char* ws = (char*)d_ws;
  size_t off = 0;
  auto carve = [&](size_t bytes) -> char* {
    char* p = ws + off;
    off += (bytes + 255) & ~(size_t)255;
    return p;
  };
  unsigned short* h1_bf  = (unsigned short*)carve((size_t)T_DIM * D_DIM * 2);
  float*          qkv    = (float*)         carve((size_t)T_DIM * QKV_N * 4);
  unsigned short* q_bf   = (unsigned short*)carve((size_t)T_DIM * H_N * HD_N * 2);
  unsigned short* k_bf   = (unsigned short*)carve((size_t)T_DIM * KH_N * HD_N * 2);
  unsigned short* v_bf   = (unsigned short*)carve((size_t)T_DIM * KH_N * HD_N * 2);
  unsigned short* o_bf   = (unsigned short*)carve((size_t)T_DIM * H_N * HD_N * 2);
  float*          a_proj = (float*)         carve((size_t)T_DIM * D_DIM * 4);
  float*          resid  = (float*)         carve((size_t)T_DIM * D_DIM * 4);
  float*          h2_f32 = (float*)         carve((size_t)T_DIM * D_DIM * 4);
  unsigned short* h2_bf  = (unsigned short*)carve((size_t)T_DIM * D_DIM * 2);
  float*          comb   = (float*)         carve((size_t)T_DIM * E_N * 4);
  float*          gu     = (float*)         carve((size_t)E_N * T_DIM * GU_N * 4);
  unsigned short* act_bf = (unsigned short*)carve((size_t)E_N * T_DIM * I_N * 2);
  float*          moe    = (float*)         carve((size_t)T_DIM * D_DIM * 4);
  (void)ws_size; (void)n_in; (void)in_sizes; (void)out_size;

  // 1) h1 = rmsnorm(hidden, w_pre_attn) -> bf16
  k_rmsnorm<<<T_DIM, 256, 0, stream>>>(hidden, w_pre_attn, nullptr, h1_bf);

  // 2) qkv = h1 @ w_qkv   (1024 x 3072 x 2048)
  k_gemm_bf16<0><<<dim3(QKV_N / BN, T_DIM / BM, 1), 256, 0, stream>>>(
      h1_bf, w_qkv, qkv, T_DIM, QKV_N, D_DIM, 0, 0, 0, nullptr, 0);

  // 3) RoPE + bf16 repack
  k_rope<<<T_DIM, 256, 0, stream>>>(positions, qkv, q_bf, k_bf, v_bf);

  // 4) flash attention (causal, softcap)
  k_attn<<<dim3(T_DIM / QT, H_N, 1), 256, 0, stream>>>(q_bf, k_bf, v_bf, o_bf);

  // 5) a = o @ w_o   (1024 x 2048 x 2048)
  k_gemm_bf16<0><<<dim3(D_DIM / BN, T_DIM / BM, 1), 256, 0, stream>>>(
      o_bf, w_o, a_proj, T_DIM, D_DIM, H_N * HD_N, 0, 0, 0, nullptr, 0);

  // 6) resid = hidden + rmsnorm(a, w_post_attn)
  k_add_rmsnorm<<<T_DIM, 256, 0, stream>>>(a_proj, w_post_attn, hidden, resid);

  // 7) h2 = rmsnorm(resid, w_pre_moe) -> fp32 + bf16
  k_rmsnorm<<<T_DIM, 256, 0, stream>>>(resid, w_pre_moe, h2_f32, h2_bf);

  // 8) router
  k_gate<<<T_DIM, 32, 0, stream>>>(h2_f32, w_gate, comb);

  // 9) gu[e] = h2 @ w_gate_up[e]   (1024 x 4096 x 2048, z = expert)
  k_gemm_bf16<0><<<dim3(GU_N / BN, T_DIM / BM, E_N), 256, 0, stream>>>(
      h2_bf, w_gate_up, gu, T_DIM, GU_N, D_DIM,
      0, (long long)D_DIM * GU_N, (long long)T_DIM * GU_N, nullptr, 0);

  // 10) act = gelu(g) * u -> bf16
  k_act<<<(unsigned)((size_t)E_N * T_DIM * I_N / 256), 256, 0, stream>>>(gu, act_bf);

  // 11) moe = 0; moe += comb[t,e] * (act[e] @ w_down[e])
  k_zero<<<(T_DIM * D_DIM) / 256, 256, 0, stream>>>(moe);
  k_gemm_bf16<1><<<dim3(D_DIM / BN, T_DIM / BM, E_N), 256, 0, stream>>>(
      act_bf, w_down, moe, T_DIM, D_DIM, I_N,
      (long long)T_DIM * I_N, (long long)I_N * D_DIM, 0, comb, E_N);

  // 12) out = resid + rmsnorm(moe, w_post_moe)
  k_add_rmsnorm<<<T_DIM, 256, 0, stream>>>(moe, w_post_moe, resid, out);
}
